// MatrixKANLayer_88098369175692
// MI455X (gfx1250) — matrix operand order repaired and checked by reading, not compiled, not run on hardware
//
#include <hip/hip_runtime.h>

// MatrixKAN layer as a dense bf16 WMMA GEMM over an expanded K dimension.
// B=8192, IN=512, OUT=512, G=20. Each input feature gets 32 K-slots:
//   slots 0..20  : spline hat weights (1-t at idx, t at idx+1)
//   slot  21     : silu(clip(x)) paired with base_w
//   slots 22..31 : zero padding (one 16x16x32 WMMA K-step == one feature)
// y = A_expanded @ W_packed^T with v_wmma_f32_16x16x32_bf16, f32 accumulate.
// This round: sched_group_barrier-enforced software pipeline — issue all 16
// B loads of feature i+1 as one group, then the 8 WMMAs of feature i, so the
// WMMAs run under outstanding loads (partial loadcnt waits, not full drains).

static constexpr int NB = 8192;         // batch
static constexpr int NI = 512;          // in features
static constexpr int NO = 512;          // out features
static constexpr int NG = 20;           // grid size
static constexpr int SLOTS = 32;        // padded slots per feature
static constexpr int NK = NI * SLOTS;   // expanded K = 16384

typedef __attribute__((ext_vector_type(16))) __bf16 v16bf;
typedef __attribute__((ext_vector_type(8)))  float  v8f;

__device__ __forceinline__ unsigned bf16b(float f) {
  // round-to-nearest-even f32 -> bf16 bit pattern
  unsigned u = __float_as_uint(f);
  return ((u + 0x7fffu + ((u >> 16) & 1u)) >> 16) & 0xffffu;
}

// ---------------------------------------------------------------------------
// Pack coeffs [NO, NI, 21] + base_w [NO, NI] into bf16 W2 [NO, NK] row-major.
// ---------------------------------------------------------------------------
__global__ __launch_bounds__(256) void prep_w(const float* __restrict__ coeffs,
                                              const float* __restrict__ base_w,
                                              unsigned short* __restrict__ W2) {
  int tid = blockIdx.x * 256 + threadIdx.x;        // NO*NK threads
  int o = tid >> 14;                               // / NK (16384)
  int k = tid & (NK - 1);
  int i = k >> 5;
  int s = k & 31;
  float v = 0.f;
  if (s < 21)       v = coeffs[(o * NI + i) * 21 + s];
  else if (s == 21) v = base_w[o * NI + i];
  W2[tid] = (unsigned short)bf16b(v);
}

// ---------------------------------------------------------------------------
// Per-(b,i) metadata: .x = {bf16 t (hi), bf16 1-t (lo)}  .y = {idx (hi), bf16 silu (lo)}
// ---------------------------------------------------------------------------
__global__ __launch_bounds__(256) void prep_m(const float* __restrict__ x,
                                              uint2* __restrict__ meta) {
  int tid = blockIdx.x * 256 + threadIdx.x;        // NB*NI threads
  float xv = x[tid];
  float xc = fminf(1.f, fmaxf(-1.f, xv));
  float sil = xc / (1.f + __expf(-xc));            // silu(xc)
  float xg = (xc + 1.f) * 10.f;                    // (xc+1)*0.5*G, in [0,20]
  int id = (int)floorf(xg);
  id = id < 0 ? 0 : (id > NG - 1 ? NG - 1 : id);
  float t = xg - (float)id;
  uint2 m;
  m.x = bf16b(1.f - t) | (bf16b(t) << 16);
  m.y = bf16b(sil) | ((unsigned)id << 16);
  meta[tid] = m;
}

// ---------------------------------------------------------------------------
// GEMM helpers
// ---------------------------------------------------------------------------
struct BTile { uint4 q[8][2]; };   // 8 N-fragments x 32 bytes each

__device__ __forceinline__ void load_btile(const unsigned short* base, BTile& b) {
#pragma unroll
  for (int nt = 0; nt < 8; ++nt) {
    const uint4* p = (const uint4*)(base + (size_t)nt * 16 * NK);  // +16 W2 rows
    b.q[nt][0] = p[0];   // K = kh*16 + 0..7
    b.q[nt][1] = p[1];   // K = kh*16 + 8..15
  }
}

__device__ __forceinline__ v16bf build_a(uint2 mm, int kb) {
  unsigned omt = mm.x & 0xffffu;   // bf16(1-t)
  unsigned tt  = mm.x >> 16;       // bf16(t)
  unsigned sil = mm.y & 0xffffu;   // bf16(silu)
  int      idx = (int)(mm.y >> 16);
  union { unsigned u[8]; v16bf v; } A;
#pragma unroll
  for (int j = 0; j < 8; ++j) {
    int s0 = kb + 2 * j;     // element 2j:   slot kh*16 + 2j
    int s1 = kb + 2 * j + 1; // element 2j+1: slot kh*16 + 2j + 1
    unsigned v0 = (s0 == idx) ? omt : (s0 == idx + 1) ? tt : (s0 == 21) ? sil : 0u;
    unsigned v1 = (s1 == idx) ? omt : (s1 == idx + 1) ? tt : (s1 == 21) ? sil : 0u;
    A.u[j] = v0 | (v1 << 16);
  }
  return A.v;
}

__device__ __forceinline__ void mma_tile(v16bf a, const BTile& b, v8f (&acc)[8]) {
#pragma unroll
  for (int nt = 0; nt < 8; ++nt) {
    union { uint4 q[2]; v16bf v; } Bf;
    Bf.q[0] = b.q[nt][0];
    Bf.q[1] = b.q[nt][1];
    acc[nt] = __builtin_amdgcn_wmma_f32_16x16x32_bf16(
        false, a, false, Bf.v, (short)0, acc[nt], false, false);
  }
}

// Pin the pipeline order for one half-iteration:
//   group 1: 17 VMEM reads (16 x b128 B-tile prefetch + 1 x b64 meta)
//   group 2: 8 WMMAs (consuming the previously loaded tile)
// VALU (A construction, addressing) is left free to fill the load shadow.
__device__ __forceinline__ void pin_pipeline_half() {
  __builtin_amdgcn_sched_group_barrier(0x020, 17, 0);  // VMEM read group
  __builtin_amdgcn_sched_group_barrier(0x008, 8, 0);   // MFMA/WMMA group
}

// ---------------------------------------------------------------------------
// GEMM: grid (NB/128, NO/128), 256 threads = 8 waves. Wave w owns rows
// [blk*128 + w*16, +16) x all 128 cols, double-buffered B tiles in registers.
// ---------------------------------------------------------------------------
__global__ __launch_bounds__(256) void kan_gemm(const unsigned short* __restrict__ W2,
                                                const uint2* __restrict__ meta,
                                                float* __restrict__ out) {
  const int lane = threadIdx.x & 31;
  const int wave = threadIdx.x >> 5;
  const int ml = lane & 15;     // M for A / N for B / N for C
  const int kh = lane >> 4;     // K-half selector
  const int m0 = blockIdx.x * 128 + wave * 16;
  const int n0 = blockIdx.y * 128;
  const int kb = kh * 16;       // A slot base for this lane half
  const int gr = m0 + ml;       // batch row this lane reads meta for

  v8f acc[8];
#pragma unroll
  for (int nt = 0; nt < 8; ++nt) acc[nt] = v8f{0.f, 0.f, 0.f, 0.f, 0.f, 0.f, 0.f, 0.f};

  const uint2* mrow = meta + (size_t)gr * NI;
  // B: lane reads row (n0+ml [+nt*16]), K-chunk [i*32 + kh*16, +16)
  const unsigned short* wbase = W2 + (size_t)(n0 + ml) * NK + kh * 16;

  BTile b0, b1;
  load_btile(wbase, b0);            // feature 0
  uint2 mc = mrow[0];

#pragma unroll 1
  for (int i = 0; i < NI; i += 2) {
    // --- even feature i: prefetch feature i+1 into b1, consume b0 ---
    uint2 mn = mrow[i + 1];
    load_btile(wbase + (size_t)(i + 1) * SLOTS, b1);
    mma_tile(build_a(mc, kb), b0, acc);
    pin_pipeline_half();

    // --- odd feature i+1: prefetch feature i+2 into b0, consume b1 ---
    int i2 = (i + 2 < NI) ? (i + 2) : (NI - 1);   // clamped redundant tail load
    mc = mrow[i2];
    load_btile(wbase + (size_t)i2 * SLOTS, b0);
    mma_tile(build_a(mn, kb), b1, acc);
    pin_pipeline_half();
  }

  // C/D layout: lane half kh -> rows M = kh*8 + v, column N = ml
#pragma unroll
  for (int nt = 0; nt < 8; ++nt) {
    int col = n0 + nt * 16 + ml;
    float* orow = out + (size_t)(m0 + kh * 8) * NO + col;
#pragma unroll
    for (int v = 0; v < 8; ++v)
      orow[(size_t)v * NO] = acc[nt][v];
  }
}

// ---------------------------------------------------------------------------
extern "C" void kernel_launch(void* const* d_in, const int* in_sizes, int n_in,
                              void* d_out, int out_size, void* d_ws, size_t ws_size,
                              hipStream_t stream) {
  const float* x      = (const float*)d_in[0];   // [NB, NI]
  const float* coeffs = (const float*)d_in[1];   // [NO, NI, 21]
  const float* base_w = (const float*)d_in[2];   // [NO, NI]
  float* out = (float*)d_out;                    // [NB, NO]

  // workspace: W2 bf16 [NO*NK] (16.8 MB) then meta uint2 [NB*NI] (33.6 MB)
  unsigned short* W2 = (unsigned short*)d_ws;
  uint2* meta = (uint2*)((char*)d_ws + (size_t)NO * NK * sizeof(unsigned short));

  prep_w<<<(NO * NK) / 256, 256, 0, stream>>>(coeffs, base_w, W2);
  prep_m<<<(NB * NI) / 256, 256, 0, stream>>>(x, meta);
  kan_gemm<<<dim3(NB / 128, NO / 128), 256, 0, stream>>>(W2, meta, out);
}
